// Rnn_61452392071388
// MI455X (gfx1250) — matrix operand order repaired and checked by reading, not compiled, not run on hardware
//
#include <hip/hip_runtime.h>
#include <hip/hip_bf16.h>

// ---------------------------------------------------------------------------
// Stacked SimpleRNN inference on MI455X (gfx1250, wave32, WMMA).
//   Phase 1: xw0 = x @ W0 + b0  (big parallel bf16-WMMA GEMM, frag layout out)
//            - vectorized b64 loads for in-bounds K chunks,
//            - branch-free clamped loads + cndmask for the K=78 tail.
//   Phase 2: 32 single-wave WGs; each owns 16 batch rows and runs ALL 4 layers
//            through the full T=512 recurrence with weights + hidden state in
//            LDS, 42 v_wmma_f32_16x16x32_bf16 per step, fp32 accumulate,
//            native-exp tanh.
//   Phase 3: dense(70->5) + softmax (VALU).
// ---------------------------------------------------------------------------

typedef __bf16 bf16;
typedef __attribute__((ext_vector_type(16))) __bf16 v16bf;
typedef __attribute__((ext_vector_type(8)))  __bf16 v8bf;
typedef __attribute__((ext_vector_type(8)))  float  v8f;
typedef __attribute__((ext_vector_type(2)))  float  f32x2;

#define T_STEPS 512
#define BATCH   512
#define FEAT    78

__device__ __forceinline__ float fast_tanh(float x) {
  // tanh(x) = 1 - 2/(exp(2x)+1); v_exp_f32 + v_rcp_f32 (TRANS, co-exec w/ WMMA)
  float e2x = __expf(2.0f * x);
  return 1.0f - 2.0f * __builtin_amdgcn_rcpf(e2x + 1.0f);
}

// A fragment (16x32 bf16, M=row) from row-major LDS buffer, row stride S halfs.
// ISA layout: lane<16 -> M=lane, K = {0..7,16..23}; lane>=16 -> same M, K+8.
__device__ __forceinline__ v16bf ldsA(const bf16* buf, int S, int kc, int lane) {
  int row = lane & 15;
  int kb  = (lane >> 4) << 3;                  // 0 or 8
  const bf16* p = buf + row * S + kc + kb;
  v8bf lo = *(const v8bf*)p;                   // K = kb .. kb+7   (ds_load_b128)
  v8bf hi = *(const v8bf*)(p + 16);            // K = kb+16..kb+23 (ds_load_b128)
  v16bf a;
#pragma unroll
  for (int i = 0; i < 8; ++i) { a[i] = lo[i]; a[i + 8] = hi[i]; }
  return a;
}

// B fragment (32x16 bf16, K x N) from transposed weights wT[Npad][Kpad] in LDS.
// ISA layout: lane = N (mod 16); K = {0..7,16..23}, +8 for lanes 16..31.
__device__ __forceinline__ v16bf ldsB(const bf16* wT, int S, int nt, int kc, int lane) {
  int n     = (lane & 15) + nt * 16;
  int kbase = ((lane >> 4) << 3) + kc;         // +0 or +8
  const bf16* p = wT + n * S + kbase;
  v8bf lo = *(const v8bf*)p;
  v8bf hi = *(const v8bf*)(p + 16);
  v16bf b;
#pragma unroll
  for (int i = 0; i < 8; ++i) { b[i] = lo[i]; b[i + 8] = hi[i]; }
  return b;
}

// Stage a weight matrix (global fp32 [Kdim][Ndim], row-major) into LDS as
// transposed, zero-padded bf16 [Np][Kp].
__device__ __forceinline__ void stage_wT(bf16* dst, int Np, int Kp,
                                         const float* src, int Kdim, int Ndim,
                                         int tid, int stride) {
  for (int i = tid; i < Np * Kp; i += stride) {
    int n = i / Kp, k = i - n * Kp;
    dst[i] = (bf16)((k < Kdim && n < Ndim) ? src[k * Ndim + n] : 0.0f);
  }
}

__device__ __forceinline__ void stage_bias(float* dst, int Np, const float* src,
                                           int H, int tid, int stride) {
  for (int i = tid; i < Np; i += stride) dst[i] = (i < H) ? src[i] : 0.0f;
}

// Branch-free guarded row read: unconditional load at clamped index, then
// v_cndmask-select zero. No EXEC manipulation, no serializing branches.
__device__ __forceinline__ bf16 ldx_guard(const float* xrow, int k) {
  float v = xrow[(k < FEAT) ? k : (FEAT - 1)];
  return (bf16)((k < FEAT) ? v : 0.0f);
}

// ---------------------------------------------------------------------------
// Phase 1: xw0[t][bt][lane][v] = (x @ W0 + b0) in WMMA C-fragment layout.
// M-tile = 16 batch rows at a fixed t (matches phase-2 tiling).
// ---------------------------------------------------------------------------
__global__ __launch_bounds__(128) void rnn_inproj_kernel(
    const float* __restrict__ x, const float* __restrict__ W0,
    const float* __restrict__ b0, float* __restrict__ xw0) {
  __shared__ alignas(16) bf16 sW0[16 * 96];   // W0^T, padded K: 78 -> 96
  __shared__ float sb0[16];
  int tid = threadIdx.x;
  stage_wT(sW0, 16, 96, W0, FEAT, 16, tid, 128);
  if (tid < 16) sb0[tid] = b0[tid];
  __syncthreads();

  int wave = tid >> 5, lane = tid & 31;
  int tile = blockIdx.x * 4 + wave;           // 16384 tiles total
  int t = tile >> 5, bt = tile & 31;
  int n0 = lane & 15;
  int row = lane & 15, kb = (lane >> 4) << 3;

  v8f acc;
#pragma unroll
  for (int v = 0; v < 8; ++v) acc[v] = sb0[n0];

  const float* xrow = x + ((size_t)(bt * 16 + row) * T_STEPS + t) * FEAT;

  // K chunks 0,1: k <= 63 < 78, fully in-bounds -> vector b64 loads (row base
  // is only 8B-aligned, so float2 is the widest provably-aligned load).
#pragma unroll
  for (int c = 0; c < 2; ++c) {
    const float* p = xrow + 32 * c + kb;
    v16bf A;
#pragma unroll
    for (int g = 0; g < 4; ++g) {
      f32x2 lo = *(const f32x2*)(p + 2 * g);        // K = kb+2g, kb+2g+1
      f32x2 hi = *(const f32x2*)(p + 16 + 2 * g);   // K = kb+16+2g, ...
      A[2 * g]     = (bf16)lo.x;  A[2 * g + 1]     = (bf16)lo.y;
      A[8 + 2 * g] = (bf16)hi.x;  A[8 + 2 * g + 1] = (bf16)hi.y;
    }
    v16bf B = ldsB(sW0, 96, 0, 32 * c, lane);
    acc = __builtin_amdgcn_wmma_f32_16x16x32_bf16(false, A, false, B,
                                                  (short)0, acc, false, false);
  }
  // K chunk 2: tail (k = 64..95). j>=8 means k>=80 -> statically zero; the
  // rest use clamped unconditional loads + select (no branches).
  {
    v16bf A;
#pragma unroll
    for (int j = 0; j < 16; ++j) {
      int k = 64 + kb + j + ((j >= 8) ? 8 : 0);
      A[j] = ldx_guard(xrow, k);
    }
    v16bf B = ldsB(sW0, 96, 0, 64, lane);
    acc = __builtin_amdgcn_wmma_f32_16x16x32_bf16(false, A, false, B,
                                                  (short)0, acc, false, false);
  }

  float* xp = xw0 + (((size_t)t * 32 + bt) * 32 + lane) * 8;
  *(float4*)(xp)     = make_float4(acc[0], acc[1], acc[2], acc[3]);
  *(float4*)(xp + 4) = make_float4(acc[4], acc[5], acc[6], acc[7]);
}

// ---------------------------------------------------------------------------
// Phase 2: fused 4-layer recurrence. One wave == one 16-row batch tile.
// ---------------------------------------------------------------------------
template <int NT, int KCW, int KCU, int SIN, int SSELF, int SW, int SU>
__device__ __forceinline__ void layer_step(
    int lane, int n0, int moff, const bf16* hin, bf16* hself,
    const bf16* WT, const bf16* UT, const float* bias,
    float* lastout, int Hout, bool dolast, int rowbase) {
  v8f acc[NT];
#pragma unroll
  for (int nt = 0; nt < NT; ++nt) {
    float b = bias[nt * 16 + n0];
#pragma unroll
    for (int v = 0; v < 8; ++v) acc[nt][v] = b;
  }
#pragma unroll
  for (int c = 0; c < KCW; ++c) {             // input projection: hin @ W
    v16bf A = ldsA(hin, SIN, 32 * c, lane);
#pragma unroll
    for (int nt = 0; nt < NT; ++nt) {
      v16bf B = ldsB(WT, SW, nt, 32 * c, lane);
      acc[nt] = __builtin_amdgcn_wmma_f32_16x16x32_bf16(
          false, A, false, B, (short)0, acc[nt], false, false);
    }
  }
#pragma unroll
  for (int c = 0; c < KCU; ++c) {             // recurrence: h_{t-1} @ U
    v16bf A = ldsA(hself, SSELF, 32 * c, lane);
#pragma unroll
    for (int nt = 0; nt < NT; ++nt) {
      v16bf B = ldsB(UT, SU, nt, 32 * c, lane);
      acc[nt] = __builtin_amdgcn_wmma_f32_16x16x32_bf16(
          false, A, false, B, (short)0, acc[nt], false, false);
    }
  }
#pragma unroll
  for (int nt = 0; nt < NT; ++nt) {
#pragma unroll
    for (int v = 0; v < 8; ++v) {
      float hv = fast_tanh(acc[nt][v]);
      int n = nt * 16 + n0, m = v + moff;
      hself[m * SSELF + n] = (bf16)hv;        // padded cols compute tanh(0)=0
      if (dolast && n < Hout)
        lastout[(size_t)(rowbase + m) * Hout + n] = hv;
    }
  }
}

__global__ __launch_bounds__(32) void rnn_recurrence_kernel(
    const float* __restrict__ xw0, const float* __restrict__ U0,
    const float* __restrict__ W1, const float* __restrict__ U1, const float* __restrict__ b1,
    const float* __restrict__ W2, const float* __restrict__ U2, const float* __restrict__ b2,
    const float* __restrict__ W3, const float* __restrict__ U3, const float* __restrict__ b3,
    float* __restrict__ h3out) {
  // Transposed, zero-padded bf16 weights (resident for all 512 steps).
  __shared__ alignas(16) bf16 sU0[16 * 32];
  __shared__ alignas(16) bf16 sW1[32 * 32];
  __shared__ alignas(16) bf16 sU1[32 * 32];
  __shared__ alignas(16) bf16 sW2[64 * 32];
  __shared__ alignas(16) bf16 sU2[64 * 64];
  __shared__ alignas(16) bf16 sW3[80 * 64];
  __shared__ alignas(16) bf16 sU3[80 * 96];
  __shared__ float sb1[32], sb2[64], sb3[80];
  // Hidden states for this wave's 16 batch rows (K-padded for WMMA).
  __shared__ alignas(16) bf16 sh0[16 * 32];
  __shared__ alignas(16) bf16 sh1[16 * 32];
  __shared__ alignas(16) bf16 sh2[16 * 64];
  __shared__ alignas(16) bf16 sh3[16 * 96];

  int lane = threadIdx.x;
  stage_wT(sU0, 16, 32, U0, 16, 16, lane, 32);
  stage_wT(sW1, 32, 32, W1, 16, 32, lane, 32);
  stage_wT(sU1, 32, 32, U1, 32, 32, lane, 32);
  stage_wT(sW2, 64, 32, W2, 32, 64, lane, 32);
  stage_wT(sU2, 64, 64, U2, 64, 64, lane, 32);
  stage_wT(sW3, 80, 64, W3, 64, 70, lane, 32);
  stage_wT(sU3, 80, 96, U3, 70, 70, lane, 32);
  stage_bias(sb1, 32, b1, 32, lane, 32);
  stage_bias(sb2, 64, b2, 64, lane, 32);
  stage_bias(sb3, 80, b3, 70, lane, 32);
  for (int i = lane; i < 16 * 32; i += 32) { sh0[i] = (bf16)0.0f; sh1[i] = (bf16)0.0f; }
  for (int i = lane; i < 16 * 64; i += 32) sh2[i] = (bf16)0.0f;
  for (int i = lane; i < 16 * 96; i += 32) sh3[i] = (bf16)0.0f;
  __syncthreads();

  int bt   = blockIdx.x;                      // 0..31: 16-row batch tile
  int n0   = lane & 15;
  int moff = (lane >> 4) << 3;

  for (int t = 0; t < T_STEPS; ++t) {
    // Layer-0 accumulator init = precomputed xw0 fragment (L2-resident).
    const float* xp = xw0 + (((size_t)t * 32 + bt) * 32 + lane) * 8;
    float4 xa = *(const float4*)(xp);
    float4 xb = *(const float4*)(xp + 4);
    if (t + 1 < T_STEPS)
      __builtin_prefetch(xp + 32 * 32 * 8, 0, 0);   // next step's slice

    // ---- Layer 0: h0 = tanh(xw0_t + h0 @ U0)   (1 WMMA)
    {
      v8f acc;
      acc[0] = xa.x; acc[1] = xa.y; acc[2] = xa.z; acc[3] = xa.w;
      acc[4] = xb.x; acc[5] = xb.y; acc[6] = xb.z; acc[7] = xb.w;
      v16bf A = ldsA(sh0, 32, 0, lane);
      v16bf B = ldsB(sU0, 32, 0, 0, lane);
      acc = __builtin_amdgcn_wmma_f32_16x16x32_bf16(false, A, false, B,
                                                    (short)0, acc, false, false);
#pragma unroll
      for (int v = 0; v < 8; ++v)
        sh0[(v + moff) * 32 + n0] = (bf16)fast_tanh(acc[v]);
    }
    // ---- Layer 1: 32 units   (4 WMMA)
    layer_step<2, 1, 1, 32, 32, 32, 32>(lane, n0, moff, sh0, sh1, sW1, sU1,
                                        sb1, nullptr, 0, false, 0);
    // ---- Layer 2: 64 units   (12 WMMA)
    layer_step<4, 1, 2, 32, 64, 32, 64>(lane, n0, moff, sh1, sh2, sW2, sU2,
                                        sb2, nullptr, 0, false, 0);
    // ---- Layer 3: 70 units   (25 WMMA); emit fp32 h_last at final step
    layer_step<5, 2, 3, 64, 96, 64, 96>(lane, n0, moff, sh2, sh3, sW3, sU3,
                                        sb3, h3out, 70, t == T_STEPS - 1,
                                        bt * 16);
  }
}

// ---------------------------------------------------------------------------
// Phase 3: logits = h3 @ Wd + bd ; softmax.  (512 rows, trivial)
// ---------------------------------------------------------------------------
__global__ __launch_bounds__(256) void dense_softmax_kernel(
    const float* __restrict__ h3, const float* __restrict__ Wd,
    const float* __restrict__ bd, float* __restrict__ out) {
  int r = blockIdx.x * blockDim.x + threadIdx.x;
  if (r >= BATCH) return;
  float lg[5];
#pragma unroll
  for (int c = 0; c < 5; ++c) lg[c] = bd[c];
  for (int k = 0; k < 70; ++k) {
    float h = h3[r * 70 + k];
#pragma unroll
    for (int c = 0; c < 5; ++c) lg[c] += h * Wd[k * 5 + c];
  }
  float m = lg[0];
#pragma unroll
  for (int c = 1; c < 5; ++c) m = fmaxf(m, lg[c]);
  float s = 0.0f;
#pragma unroll
  for (int c = 0; c < 5; ++c) { lg[c] = __expf(lg[c] - m); s += lg[c]; }
  float inv = __builtin_amdgcn_rcpf(s);
#pragma unroll
  for (int c = 0; c < 5; ++c) out[r * 5 + c] = lg[c] * inv;
}

// ---------------------------------------------------------------------------
extern "C" void kernel_launch(void* const* d_in, const int* in_sizes, int n_in,
                              void* d_out, int out_size, void* d_ws, size_t ws_size,
                              hipStream_t stream) {
  (void)in_sizes; (void)n_in; (void)out_size; (void)ws_size;
  const float* x  = (const float*)d_in[0];
  const float* W0 = (const float*)d_in[1];
  const float* U0 = (const float*)d_in[2];
  const float* b0 = (const float*)d_in[3];
  const float* W1 = (const float*)d_in[4];
  const float* U1 = (const float*)d_in[5];
  const float* b1 = (const float*)d_in[6];
  const float* W2 = (const float*)d_in[7];
  const float* U2 = (const float*)d_in[8];
  const float* b2 = (const float*)d_in[9];
  const float* W3 = (const float*)d_in[10];
  const float* U3 = (const float*)d_in[11];
  const float* b3 = (const float*)d_in[12];
  const float* Wd = (const float*)d_in[13];
  const float* bd = (const float*)d_in[14];

  float* xw0 = (float*)d_ws;                               // 16.78 MB, L2-resident
  float* h3o = (float*)((char*)d_ws + (size_t)BATCH * T_STEPS * 16 * sizeof(float));

  rnn_inproj_kernel<<<4096, 128, 0, stream>>>(x, W0, b0, xw0);
  rnn_recurrence_kernel<<<32, 32, 0, stream>>>(xw0, U0, W1, U1, b1,
                                               W2, U2, b2, W3, U3, b3, h3o);
  dense_softmax_kernel<<<2, 256, 0, stream>>>(h3o, Wd, bd, (float*)d_out);
}
